// ContextGatedTokenizer_592705487402
// MI455X (gfx1250) — compile-verified, matrix operand order checked by reading, not run on hardware
//
#include <hip/hip_runtime.h>
#include <hip/hip_bf16.h>

// ---------- problem constants (from reference) ----------
constexpr int Bn   = 32;    // batch
constexpr int KQn  = 8;     // query tokens == heads
constexpr int Dn   = 256;   // head dim
constexpr int Sn   = 2048;  // history length
constexpr int KVHn = 2;     // kv heads
constexpr int Gn   = 4;     // GQA group (KQn/KVHn)
constexpr int SBLKn = 8;            // S splits per (b,h)
constexpr int SWn   = Sn / SBLKn;   // 256 s-rows per WG
constexpr int LP    = 264;          // padded bf16 row stride (elements) -> 528B, conflict free
constexpr int FLP   = 260;          // padded fp32 row stride for TDM tile (260*4B = 1040B)

// ---------- LDS layouts (ushort indices) ----------
constexpr int FSZ_US = 16 * FLP * 2;        // one fp32 hist tile = 8320 ushorts
// k_scores:
constexpr int W_OFF  = 0;                   // weights bf16 [256][LP]
constexpr int KC_OFF = Dn * LP;             // k-chunk bf16 [16][LP]            = 67584
constexpr int QP_OFF = KC_OFF + 16 * LP;    // q' bf16 [16][LP] (rows>=4 zero)  = 71808
constexpr int SQ_OFF = QP_OFF + 16 * LP;    // float region: ssq partials 8x16  = 76032
constexpr int H_OFF  = SQ_OFF + 256;        // hist chunk bf16 [16][LP]         = 76288
constexpr int F0_OFF = H_OFF + 16 * LP;     // fp32 TDM tile ping               = 80512
constexpr int F1_OFF = F0_OFF + FSZ_US;     // fp32 TDM tile pong               = 88832
constexpr size_t K2_LDS = (size_t)(F1_OFF + FSZ_US) * 2;   // 194304 B
// k_out:
constexpr int A4_OFF  = Dn * LP;            // attn floats [4][16]              = 67584
constexpr int H4_OFF  = A4_OFF + 128;       // hist chunk bf16 [16][LP]         = 67712
constexpr int F0_OFF4 = H4_OFF + 16 * LP;   // fp32 TDM tile ping               = 71936
constexpr int F1_OFF4 = F0_OFF4 + FSZ_US;   // fp32 TDM tile pong               = 80256
constexpr size_t K4_LDS = (size_t)(F1_OFF4 + FSZ_US) * 2;  // 177152 B

typedef __attribute__((ext_vector_type(16))) __bf16 v16bf;
typedef __attribute__((ext_vector_type(8)))  float  v8f;
typedef __attribute__((ext_vector_type(4)))  unsigned int v4u;
typedef __attribute__((ext_vector_type(8)))  int v8i;
typedef __attribute__((ext_vector_type(4)))  int v4i;

#if defined(__has_builtin)
#  if __has_builtin(__builtin_amdgcn_tensor_load_to_lds) && __has_builtin(__builtin_amdgcn_s_wait_tensorcnt)
#    define HAS_TDM 1
#  endif
#endif
#ifndef HAS_TDM
#  define HAS_TDM 0
#endif

__device__ __forceinline__ unsigned short f2bf(float f) {
  unsigned int u = __float_as_uint(f);
  u += 0x7FFFu + ((u >> 16) & 1u);   // round-to-nearest-even
  return (unsigned short)(u >> 16);
}

// 16x32 bf16 A/B fragment slice from a bf16 LDS row: p = &row[32*ds + 8*hi]
__device__ __forceinline__ v16bf loadFrag(const unsigned short* p) {
  union { v16bf v; uint4 q[2]; } u;
  u.q[0] = *(const uint4*)(p);
  u.q[1] = *(const uint4*)(p + 16);
  return u.v;
}

#if HAS_TDM
// TDM: DMA one 16x256 fp32 tile (row-major, stride 256) -> LDS with +4-DWORD row
// padding (pad_interval=256 DW, pad_amount=4 DW) => LDS row stride = 260 floats.
__device__ __forceinline__ void tdm_load_tile(const float* gsrc, float* ldst) {
  const unsigned long long ga = (unsigned long long)(uintptr_t)gsrc;
  v4u g0;
  g0.x = 1u;                                                // count=1 (valid descriptor)
  g0.y = (unsigned)(uintptr_t)ldst;                         // lds_addr (bytes)
  g0.z = (unsigned)ga;                                      // global_addr[31:0]
  g0.w = (unsigned)((ga >> 32) & 0x01FFFFFFu) | (2u << 30); // global_addr[56:32] | type=2
  v8i g1;
  g1[0] = (int)((2u << 16) | (1u << 20) | (7u << 22) | (3u << 25)); // 4B elems, pad 4DW per 256DW
  g1[1] = (int)(256u << 16);                                // tensor_dim0 = 256
  g1[2] = (int)(16u << 16);                                 // tensor_dim1 = 16
  g1[3] = (int)(256u << 16);                                // tile_dim0 = 256
  g1[4] = 16;                                               // tile_dim1 = 16
  g1[5] = 256;                                              // tensor_dim0_stride = 256
  g1[6] = 0;
  g1[7] = 0;
  v4i z4 = {0, 0, 0, 0};
#if __clang_major__ >= 23
  v8i z8 = {0, 0, 0, 0, 0, 0, 0, 0};
  __builtin_amdgcn_tensor_load_to_lds(g0, g1, z4, z4, z8, 0);
#else
  __builtin_amdgcn_tensor_load_to_lds(g0, g1, z4, z4, 0);
#endif
}
#endif

// ---------------- K1: q' = rmsnorm(target@wq^T)*(1+gq)*(1+gk)  (exact fp32) ----------------
__global__ void __launch_bounds__(256) k_qproj(const float* __restrict__ tgt,
                                               const float* __restrict__ wq,
                                               const float* __restrict__ gq,
                                               const float* __restrict__ gk,
                                               float* __restrict__ qp) {
  __shared__ float t[Dn];
  __shared__ float red[Dn];
  const int row = blockIdx.x;           // b*8 + k
  const int e = threadIdx.x;
  t[e] = tgt[(size_t)row * Dn + e];
  __syncthreads();
  const float* wr = wq + (size_t)e * Dn;
  float acc = 0.f;
#pragma unroll 4
  for (int d = 0; d < Dn; ++d) acc = fmaf(t[d], wr[d], acc);
  red[e] = acc * acc;
  __syncthreads();
  for (int s = 128; s > 0; s >>= 1) {
    if (e < s) red[e] += red[e + s];
    __syncthreads();
  }
  const float rs = rsqrtf(red[0] * (1.f / 256.f) + 1e-6f);
  qp[(size_t)row * Dn + e] = acc * rs * (1.f + gq[e]) * (1.f + gk[e]);
}

// ---------------- K2: k-projection (bf16 WMMA) + raw scores ----------------
__global__ void __launch_bounds__(256) k_scores(const float* __restrict__ hist,
                                                const float* __restrict__ wk,
                                                const float* __restrict__ qp,
                                                const unsigned char* __restrict__ mask,
                                                float* __restrict__ scoresG) {
  extern __shared__ unsigned short sm[];
  float* sqp = (float*)(sm + SQ_OFF);   // [wave][16] ssq partials

  const int wg = blockIdx.x;            // 0..511
  const int sb = wg & (SBLKn - 1);
  const int bh = wg >> 3;
  const int h  = bh & (KVHn - 1);
  const int b  = bh >> 1;
  const int tid = threadIdx.x;
  const int lane = tid & 31, wv = tid >> 5;
  const int li = lane & 15, hi = lane >> 4;

  // weights (head h) -> LDS bf16, padded rows
  const float* wkh = wk + (size_t)h * Dn * Dn;
  for (int idx = tid; idx < Dn * Dn; idx += 256)
    sm[W_OFF + (idx >> 8) * LP + (idx & 255)] = f2bf(wkh[idx]);
  // q' rows (g = 0..3) padded to 16 zero rows
  for (int idx = tid; idx < 16 * LP; idx += 256) sm[QP_OFF + idx] = 0;
  __syncthreads();
  const float* qpb = qp + ((size_t)b * KQn + h * Gn) * Dn;
  for (int idx = tid; idx < Gn * Dn; idx += 256)
    sm[QP_OFF + (idx >> 8) * LP + (idx & 255)] = f2bf(qpb[idx]);

  const float* hb = hist + ((size_t)b * Sn + sb * SWn) * Dn;
  float* scb = scoresG + ((size_t)(b * KQn + h * Gn)) * Sn + sb * SWn;
  const unsigned char* mrow = mask + (size_t)b * Sn + sb * SWn;

  const int e0 = 32 * wv, e1 = e0 + 16;
  constexpr int NCH = SWn / 16;

#if HAS_TDM
  if (wv == 0) tdm_load_tile(hb, (float*)(sm + F0_OFF));   // prefetch chunk 0
#endif

  for (int ch = 0; ch < NCH; ++ch) {
#if HAS_TDM
    const float* fbuf = (const float*)(sm + ((ch & 1) ? F1_OFF : F0_OFF));
    if (wv == 0) {
      __builtin_amdgcn_s_wait_tensorcnt(0);                // chunk ch landed in LDS
      if (ch + 1 < NCH)                                    // overlap DMA of ch+1 with compute
        tdm_load_tile(hb + (size_t)(ch + 1) * 16 * Dn,
                      (float*)(sm + ((ch & 1) ? F0_OFF : F1_OFF)));
    }
    __syncthreads();
    // cooperative fp32 -> bf16 conversion (once per chunk, paired stores)
    for (int idx = tid; idx < 16 * 128; idx += 256) {
      const int r = idx >> 7, c = (idx & 127) * 2;
      const float2 f = *(const float2*)&fbuf[r * FLP + c];
      const unsigned u = (unsigned)f2bf(f.x) | ((unsigned)f2bf(f.y) << 16);
      *(unsigned*)&sm[H_OFF + r * LP + c] = u;
    }
    __syncthreads();
#else
    for (int idx = tid; idx < 16 * Dn; idx += 256)
      sm[H_OFF + (idx >> 8) * LP + (idx & 255)] =
          f2bf(hb[(size_t)(ch * 16 + (idx >> 8)) * Dn + (idx & 255)]);
    __syncthreads();
#endif

    v8f c0 = {0.f,0.f,0.f,0.f,0.f,0.f,0.f,0.f};
    v8f c1 = {0.f,0.f,0.f,0.f,0.f,0.f,0.f,0.f};
#pragma unroll
    for (int ds = 0; ds < 8; ++ds) {
      const int kb = 32 * ds + 8 * hi;
      v16bf a  = loadFrag(&sm[H_OFF + li * LP + kb]);
      v16bf b0 = loadFrag(&sm[W_OFF + (e0 + li) * LP + kb]);
      v16bf b1 = loadFrag(&sm[W_OFF + (e1 + li) * LP + kb]);
      c0 = __builtin_amdgcn_wmma_f32_16x16x32_bf16(false, a, false, b0, (short)0, c0, false, false);
      c1 = __builtin_amdgcn_wmma_f32_16x16x32_bf16(false, a, false, b1, (short)0, c1, false, false);
    }

    // per-row sum of squares: deterministic lane-butterfly then per-wave slot
    float p[8];
#pragma unroll
    for (int r = 0; r < 8; ++r) p[r] = c0[r] * c0[r] + c1[r] * c1[r];
#pragma unroll
    for (int m2 = 1; m2 < 16; m2 <<= 1) {
#pragma unroll
      for (int r = 0; r < 8; ++r) p[r] += __shfl_xor(p[r], m2, 32);
    }
    if (li == 0) {
#pragma unroll
      for (int r = 0; r < 8; ++r) sqp[wv * 16 + r + 8 * hi] = p[r];
    }
    // spill raw k chunk as bf16 [s][e]
#pragma unroll
    for (int r = 0; r < 8; ++r) {
      sm[KC_OFF + (r + 8 * hi) * LP + e0 + li] = f2bf(c0[r]);
      sm[KC_OFF + (r + 8 * hi) * LP + e1 + li] = f2bf(c1[r]);
    }
    __syncthreads();

    if (wv == 0) {  // scores: D[g,s] = sum_e q'[g,e] * k_raw[s,e]
      v8f sc = {0.f,0.f,0.f,0.f,0.f,0.f,0.f,0.f};
#pragma unroll
      for (int ds = 0; ds < 8; ++ds) {
        const int kb = 32 * ds + 8 * hi;
        v16bf a  = loadFrag(&sm[QP_OFF + li * LP + kb]);   // rows >= 4 are zero
        v16bf b2 = loadFrag(&sm[KC_OFF + li * LP + kb]);
        sc = __builtin_amdgcn_wmma_f32_16x16x32_bf16(false, a, false, b2, (short)0, sc, false, false);
      }
      const int s = li;
      float ss = 0.f;
#pragma unroll
      for (int w = 0; w < 8; ++w) ss += sqp[w * 16 + s];   // fixed order
      const float rsb = rsqrtf(ss * (1.f / 256.f) + 1e-6f);
      const bool msk = mrow[ch * 16 + s] != 0;
      if (hi == 0) {
#pragma unroll
        for (int g = 0; g < 4; ++g) {
          float v = msk ? -__builtin_huge_valf() : sc[g] * rsb * 0.0625f;
          scb[(size_t)g * Sn + ch * 16 + s] = v;
        }
      }
    }
    __syncthreads();
  }
}

// ---------------- K3: softmax over S per (b,k) row ----------------
__global__ void __launch_bounds__(256) k_softmax(const float* __restrict__ sc,
                                                 float* __restrict__ attnOut,
                                                 float* __restrict__ attnWs) {
  __shared__ float red[256];
  const int row = blockIdx.x, tid = threadIdx.x;
  const float* src = sc + (size_t)row * Sn;
  float v[8];
  float m = -__builtin_huge_valf();
#pragma unroll
  for (int i = 0; i < 8; ++i) { v[i] = src[tid + 256 * i]; m = fmaxf(m, v[i]); }
  red[tid] = m; __syncthreads();
  for (int s = 128; s > 0; s >>= 1) { if (tid < s) red[tid] = fmaxf(red[tid], red[tid + s]); __syncthreads(); }
  m = red[0]; __syncthreads();
  float sum = 0.f;
#pragma unroll
  for (int i = 0; i < 8; ++i) { v[i] = expf(v[i] - m); sum += v[i]; }
  red[tid] = sum; __syncthreads();
  for (int s = 128; s > 0; s >>= 1) { if (tid < s) red[tid] += red[tid + s]; __syncthreads(); }
  const float inv = 1.f / red[0];
  float* d1 = attnOut + (size_t)row * Sn;
  float* d2 = attnWs + (size_t)row * Sn;
#pragma unroll
  for (int i = 0; i < 8; ++i) { float a = v[i] * inv; d1[tid + 256 * i] = a; d2[tid + 256 * i] = a; }
}

// ---------------- K4: v-projection (bf16 WMMA) + attn-weighted accumulate ----------------
__global__ void __launch_bounds__(256) k_out(const float* __restrict__ hist,
                                             const float* __restrict__ wvw,
                                             const float* __restrict__ attnG,
                                             float* __restrict__ part) {
  extern __shared__ unsigned short sm[];
  float* attnL = (float*)(sm + A4_OFF);  // [4][16]

  const int wg = blockIdx.x;
  const int sb = wg & (SBLKn - 1);
  const int bh = wg >> 3;
  const int h  = bh & (KVHn - 1);
  const int b  = bh >> 1;
  const int tid = threadIdx.x;
  const int lane = tid & 31, wv = tid >> 5;
  const int li = lane & 15, hi = lane >> 4;

  const float* wvh = wvw + (size_t)h * Dn * Dn;
  for (int idx = tid; idx < Dn * Dn; idx += 256)
    sm[W_OFF + (idx >> 8) * LP + (idx & 255)] = f2bf(wvh[idx]);

  float acc[4][2] = {{0.f,0.f},{0.f,0.f},{0.f,0.f},{0.f,0.f}};
  const float* hb = hist + ((size_t)b * Sn + sb * SWn) * Dn;
  const float* ab = attnG + ((size_t)(b * KQn + h * Gn)) * Sn + sb * SWn;
  const int e0 = 32 * wv, e1 = e0 + 16;
  constexpr int NCH = SWn / 16;

#if HAS_TDM
  if (wv == 0) tdm_load_tile(hb, (float*)(sm + F0_OFF4));  // prefetch chunk 0
#endif

  for (int ch = 0; ch < NCH; ++ch) {
    if (tid < 64) attnL[tid] = ab[(size_t)(tid >> 4) * Sn + ch * 16 + (tid & 15)];
#if HAS_TDM
    const float* fbuf = (const float*)(sm + ((ch & 1) ? F1_OFF4 : F0_OFF4));
    if (wv == 0) {
      __builtin_amdgcn_s_wait_tensorcnt(0);
      if (ch + 1 < NCH)
        tdm_load_tile(hb + (size_t)(ch + 1) * 16 * Dn,
                      (float*)(sm + ((ch & 1) ? F0_OFF4 : F1_OFF4)));
    }
    __syncthreads();
    for (int idx = tid; idx < 16 * 128; idx += 256) {
      const int r = idx >> 7, c = (idx & 127) * 2;
      const float2 f = *(const float2*)&fbuf[r * FLP + c];
      const unsigned u = (unsigned)f2bf(f.x) | ((unsigned)f2bf(f.y) << 16);
      *(unsigned*)&sm[H4_OFF + r * LP + c] = u;
    }
    __syncthreads();
#else
    for (int idx = tid; idx < 16 * Dn; idx += 256)
      sm[H4_OFF + (idx >> 8) * LP + (idx & 255)] =
          f2bf(hb[(size_t)(ch * 16 + (idx >> 8)) * Dn + (idx & 255)]);
    __syncthreads();
#endif

    v8f c0 = {0.f,0.f,0.f,0.f,0.f,0.f,0.f,0.f};
    v8f c1 = {0.f,0.f,0.f,0.f,0.f,0.f,0.f,0.f};
#pragma unroll
    for (int ds = 0; ds < 8; ++ds) {
      const int kb = 32 * ds + 8 * hi;
      v16bf a  = loadFrag(&sm[H4_OFF + li * LP + kb]);
      v16bf b0 = loadFrag(&sm[W_OFF + (e0 + li) * LP + kb]);
      v16bf b1 = loadFrag(&sm[W_OFF + (e1 + li) * LP + kb]);
      c0 = __builtin_amdgcn_wmma_f32_16x16x32_bf16(false, a, false, b0, (short)0, c0, false, false);
      c1 = __builtin_amdgcn_wmma_f32_16x16x32_bf16(false, a, false, b1, (short)0, c1, false, false);
    }
#pragma unroll
    for (int r = 0; r < 8; ++r) {
      const int srow = r + 8 * hi;
#pragma unroll
      for (int g = 0; g < 4; ++g) {
        const float w = attnL[g * 16 + srow];   // LDS broadcast
        acc[g][0] = fmaf(w, c0[r], acc[g][0]);
        acc[g][1] = fmaf(w, c1[r], acc[g][1]);
      }
    }
    __syncthreads();
  }
  // combine the two 16-lane halves (different s rows, same e)
#pragma unroll
  for (int g = 0; g < 4; ++g) {
#pragma unroll
    for (int t = 0; t < 2; ++t) {
      float v2 = acc[g][t] + __shfl_xor(acc[g][t], 16, 32);
      if (hi == 0) {
        const int e = 32 * wv + 16 * t + li;
        part[((size_t)sb * 64 + b * 2 + h) * 1024 + g * 256 + e] = v2;
      }
    }
  }
}

// ---------------- K5: sum S-split partials (deterministic order) ----------------
__global__ void __launch_bounds__(256) k_reduce(const float* __restrict__ part,
                                                float* __restrict__ out) {
  const int i = blockIdx.x * 256 + threadIdx.x;  // 0..65535
  float s = 0.f;
#pragma unroll
  for (int sb = 0; sb < SBLKn; ++sb) s += part[(size_t)sb * 65536 + i];
  out[i] = s;
}

extern "C" void kernel_launch(void* const* d_in, const int* in_sizes, int n_in,
                              void* d_out, int out_size, void* d_ws, size_t ws_size,
                              hipStream_t stream) {
  (void)in_sizes; (void)n_in; (void)out_size; (void)ws_size;
  const float* tgt  = (const float*)d_in[0];
  const float* hist = (const float*)d_in[1];
  const unsigned char* mask = (const unsigned char*)d_in[2];
  const float* wq = (const float*)d_in[3];
  const float* wk = (const float*)d_in[4];
  const float* wvw = (const float*)d_in[5];
  const float* gq = (const float*)d_in[6];
  const float* gk = (const float*)d_in[7];

  float* out     = (float*)d_out;                 // [B,K,D] = 65536
  float* attnOut = out + Bn * KQn * Dn;           // [B,K,1,S] = 524288
  float* ws      = (float*)d_ws;
  float* qp      = ws;                            // 65536 floats
  float* scores  = ws + Bn * KQn * Dn;            // 524288 floats
  float* partial = scores + Bn * KQn * Sn;        // 8*65536 floats

  // allow >64KB dynamic LDS (CDNA5: up to 320KB per workgroup)
  hipFuncSetAttribute(reinterpret_cast<const void*>(k_scores),
                      hipFuncAttributeMaxDynamicSharedMemorySize, (int)K2_LDS);
  hipFuncSetAttribute(reinterpret_cast<const void*>(k_out),
                      hipFuncAttributeMaxDynamicSharedMemorySize, (int)K4_LDS);

  k_qproj<<<Bn * KQn, 256, 0, stream>>>(tgt, wq, gq, gk, qp);
  k_scores<<<Bn * KVHn * SBLKn, 256, K2_LDS, stream>>>(hist, wk, qp, mask, scores);
  k_softmax<<<Bn * KQn, 256, 0, stream>>>(scores, attnOut, scores);
  k_out<<<Bn * KVHn * SBLKn, 256, K4_LDS, stream>>>(hist, wvw, scores, partial);
  k_reduce<<<Bn * KQn * Dn / 256, 256, 0, stream>>>(partial, out);
}
